// SkeletonLoss_16329465659750
// MI455X (gfx1250) — compile-verified, matrix operand order checked by reading, not run on hardware
//
#include <hip/hip_runtime.h>
#include <hip/hip_bf16.h>
#include <stdint.h>

typedef __attribute__((ext_vector_type(16))) _Float16 v16h;
typedef __attribute__((ext_vector_type(8)))  float    v8f;

#if __has_builtin(__builtin_amdgcn_global_load_async_to_lds_b64) && \
    __has_builtin(__builtin_amdgcn_s_wait_asynccnt)
#define HAVE_ASYNC_LDS 1
typedef int v2i __attribute__((vector_size(2 * sizeof(int))));
#endif

#define IMG_H 1024
#define IMG_W 1024
#define NBATCH 32
#define NPIX (NBATCH * IMG_H * IMG_W)          // 33554432
#define NTILES (NBATCH * 64 * 64)              // 131072 (16x16 tiles)
#define WPB 8                                  // waves per block
#define TILE_BLOCKS (NTILES / WPB)             // 16384

// ---------------------------------------------------------------- init ctl
// ctl layout (u32): [0..9] sumsP per iter, [10..19] sumsT, [20] doneP, [21] doneT
__global__ void init_ctl_k(unsigned* ctl) {
    if (threadIdx.x < 32) ctl[threadIdx.x] = 0u;
}

// ---------------------------------------------------------------- binarize
// sigmoid(pred) > 0.5  <=>  pred > 0 ;  target > 0.5
__global__ __launch_bounds__(256) void binarize_k(
    const float* __restrict__ pred, const float* __restrict__ tgt,
    uint8_t* __restrict__ binP, uint8_t* __restrict__ binT,
    uint8_t* __restrict__ skP,  uint8_t* __restrict__ skT)
{
    size_t i = (size_t)blockIdx.x * 256 + threadIdx.x;
    uint8_t bp = (pred[i] > 0.0f) ? 1 : 0;
    uint8_t bt = (tgt[i]  > 0.5f) ? 1 : 0;
    binP[i] = bp; skP[i] = bp;
    binT[i] = bt; skT[i] = bt;
}

// f16 16x32 A-fragment layout helper: lane L holds row M=L%16; VGPR v holds
// K pair with base = (v<4 ? 2v : 16+2(v-4)) + 8*(L>=16). B mirrors with N=L%16.
__device__ __forceinline__ int frag_kbase(int v, int half) {
    return (v < 4 ? 2 * v : 16 + 2 * (v - 4)) + 8 * half;
}

// ---------------------------------------------------------------- erosion step
// One wave per 16x16 output tile; 18-row u8 halo staged in LDS (stride 32,
// X-column cc lives at byte cc+7). Vertical 3-tap stage is one
// V_WMMA_F32_16X16X32_F16:  O = A(banded 0/1) x B(horizontal 3-tap row sums).
__global__ __launch_bounds__(256) void erode_k(
    const uint8_t* __restrict__ src, uint8_t* __restrict__ dst,
    unsigned* __restrict__ sumSlot, const unsigned* __restrict__ doneFlag)
{
    __shared__ uint8_t lds[WPB * 576];
    const int tid  = threadIdx.x;
    const int lane = tid & 31;
    const int wv   = tid >> 5;
    const int tile = blockIdx.x * WPB + wv;
    const int img  = tile >> 12;
    const int t    = tile & 4095;
    const int ty   = t >> 6, tx = t & 63;
    const int r0   = ty * 16, c0 = tx * 16;
    const size_t base = (size_t)img << 20;

    const unsigned done = *doneFlag;       // uniform across whole grid
    if (done) {                            // frozen: keep ping-pong parity valid
        for (int idx = lane; idx < 256; idx += 32) {
            int rr = idx >> 4, cc = idx & 15;
            size_t g = base + (size_t)(r0 + rr) * IMG_W + (c0 + cc);
            dst[g] = src[g];
        }
        return;
    }

    __builtin_prefetch(src + base + (size_t)r0 * IMG_W + c0, 0, 1);

    uint8_t* T = &lds[wv * 576];           // 18 rows, stride 32
#if HAVE_ASYNC_LDS
    const bool interior = (tx >= 1) & (tx <= 62) & (ty >= 1) & (ty <= 62);
    if (interior) {
        // Aligned 32B span [c0-8, c0+24) covers the 18B halo; 4 x b64 per row.
        const uint8_t* g0 = src + base + (size_t)(r0 - 1) * IMG_W + (c0 - 8);
        for (int i = lane; i < 72; i += 32) {
            int row = i >> 2, seg = (i & 3) * 8;
            __builtin_amdgcn_global_load_async_to_lds_b64(
                (v2i*)(g0 + (size_t)row * IMG_W + seg),
                (v2i*)(T + row * 32 + seg), 0, 0);
        }
        __builtin_amdgcn_s_wait_asynccnt(0);
    } else
#endif
    {
        for (int i = lane; i < 18 * 18; i += 32) {
            int rr = i / 18, cc = i % 18;
            int gr = r0 - 1 + rr, gc = c0 - 1 + cc;
            uint8_t v = 0;
            if (gr >= 0 && gr < IMG_H && gc >= 0 && gc < IMG_W)
                v = src[base + (size_t)gr * IMG_W + gc];
            T[rr * 32 + cc + 7] = v;
        }
    }
    __syncthreads();

    const int N    = lane & 15;
    const int half = lane >> 4;

    v16h a, b;
#pragma unroll
    for (int v = 0; v < 8; ++v) {
        int kb = frag_kbase(v, half);
#pragma unroll
        for (int p = 0; p < 2; ++p) {
            int k = kb + p;
            a[2 * v + p] = (k >= N && k <= N + 2) ? (_Float16)1.0f : (_Float16)0.0f;
            unsigned rs = 0;
            if (k < 18)
                rs = (unsigned)T[k * 32 + N + 7] + T[k * 32 + N + 8] + T[k * 32 + N + 9];
            b[2 * v + p] = (_Float16)(float)rs;
        }
    }

    v8f c = {};
    c = __builtin_amdgcn_wmma_f32_16x16x32_f16(false, a, false, b,
                                               (short)0, c, false, false);

    // eroded=(box>=9); diff=skel-eroded; skel' = skel*(diff<0.5)+eroded
    unsigned wsum = 0;
#pragma unroll
    for (int dv = 0; dv < 8; ++dv) {
        int orow = dv + 8 * half;          // D layout: M = vgpr + 8*(lane>=16)
        int e    = (c[dv] >= 8.5f) ? 1 : 0;
        int old  = (int)T[(orow + 1) * 32 + N + 8];
        int diff = old - e;
        int ns   = ((diff <= 0) ? old : 0) + e;
        dst[base + (size_t)(r0 + orow) * IMG_W + (c0 + N)] = (uint8_t)ns;
        wsum += (unsigned)ns;
    }
    for (int o = 16; o; o >>= 1) wsum += __shfl_xor(wsum, o, 32);
    if (lane == 0) atomicAdd(sumSlot, wsum);   // integer: order-independent
}

// ---------------------------------------------------------------- convergence flag
__global__ void flag_k(unsigned* ctl, int it) {
    if (threadIdx.x == 0) {
        if (!ctl[20] && it > 0 && ctl[it] == ctl[it - 1]) ctl[20] = 1;
    } else if (threadIdx.x == 1) {
        if (!ctl[21] && it > 0 && ctl[10 + it] == ctl[10 + it - 1]) ctl[21] = 1;
    }
}

// ---------------------------------------------------------------- sobel + combine
// One wave per 16x16 tile. Separable Sobel: horizontal taps in VALU from LDS,
// vertical taps as banded-matrix WMMAs (A fragments shared by pred/target):
//   gx = Av(1,2,1) x DH,  gy = Av(-1,0,1) x SH        (exact small ints)
__global__ __launch_bounds__(256) void final_k(
    const uint8_t* __restrict__ binP, const uint8_t* __restrict__ binT,
    const uint8_t* __restrict__ skP,  const uint8_t* __restrict__ skT,
    float* __restrict__ pI, float* __restrict__ pP, float* __restrict__ pT)
{
    __shared__ uint8_t lds[WPB * 768];     // per wave: two 18x20 halo regions
    const int tid  = threadIdx.x;
    const int lane = tid & 31;
    const int wv   = tid >> 5;
    const int tile = blockIdx.x * WPB + wv;
    const int img  = tile >> 12;
    const int t    = tile & 4095;
    const int r0   = (t >> 6) * 16, c0 = (t & 63) * 16;
    const size_t base = (size_t)img << 20;

    uint8_t* TP = &lds[wv * 768];
    uint8_t* TT = TP + 384;
    for (int i = lane; i < 18 * 18; i += 32) {
        int rr = i / 18, cc = i % 18;
        int gr = r0 - 1 + rr, gc = c0 - 1 + cc;
        uint8_t vp = 0, vt = 0;
        if (gr >= 0 && gr < IMG_H && gc >= 0 && gc < IMG_W) {
            size_t g = base + (size_t)gr * IMG_W + gc;
            vp = binP[g]; vt = binT[g];
        }
        TP[rr * 20 + cc] = vp;
        TT[rr * 20 + cc] = vt;
    }
    __syncthreads();

    const int N    = lane & 15;
    const int half = lane >> 4;

    v16h aS, aD, bDP, bSP, bDT, bST;
#pragma unroll
    for (int v = 0; v < 8; ++v) {
        int kb = frag_kbase(v, half);
#pragma unroll
        for (int p = 0; p < 2; ++p) {
            int k = kb + p, d = k - N;
            aS[2 * v + p] = (_Float16)(float)((d == 0 || d == 2) ? 1 : (d == 1 ? 2 : 0));
            aD[2 * v + p] = (_Float16)(float)(d == 0 ? -1 : (d == 2 ? 1 : 0));
            int dhp = 0, shp = 0, dht = 0, sht = 0;
            if (k < 18) {
                int p0 = TP[k * 20 + N], p1 = TP[k * 20 + N + 1], p2 = TP[k * 20 + N + 2];
                int t0 = TT[k * 20 + N], t1 = TT[k * 20 + N + 1], t2 = TT[k * 20 + N + 2];
                dhp = p2 - p0; shp = p0 + 2 * p1 + p2;
                dht = t2 - t0; sht = t0 + 2 * t1 + t2;
            }
            bDP[2 * v + p] = (_Float16)(float)dhp;
            bSP[2 * v + p] = (_Float16)(float)shp;
            bDT[2 * v + p] = (_Float16)(float)dht;
            bST[2 * v + p] = (_Float16)(float)sht;
        }
    }

    v8f z = {};
    v8f gxP = __builtin_amdgcn_wmma_f32_16x16x32_f16(false, aS, false, bDP, (short)0, z, false, false);
    v8f gyP = __builtin_amdgcn_wmma_f32_16x16x32_f16(false, aD, false, bSP, (short)0, z, false, false);
    v8f gxT = __builtin_amdgcn_wmma_f32_16x16x32_f16(false, aS, false, bDT, (short)0, z, false, false);
    v8f gyT = __builtin_amdgcn_wmma_f32_16x16x32_f16(false, aD, false, bST, (short)0, z, false, false);

    float aI = 0.f, aP = 0.f, aT = 0.f;
#pragma unroll
    for (int dv = 0; dv < 8; ++dv) {
        int orow = dv + 8 * half;
        size_t g = base + (size_t)(r0 + orow) * IMG_W + (c0 + N);
        float ep = sqrtf(gxP[dv] * gxP[dv] + gyP[dv] * gyP[dv] + 1e-8f);
        float et = sqrtf(gxT[dv] * gxT[dv] + gyT[dv] * gyT[dv] + 1e-8f);
        float vp = fminf(fmaxf((float)skP[g] + 0.5f * ep, 0.0f), 1.0f);
        float vt = fminf(fmaxf((float)skT[g] + 0.5f * et, 0.0f), 1.0f);
        aI += vp * vt; aP += vp; aT += vt;
    }
    for (int o = 16; o; o >>= 1) {         // fixed-order wave reduction
        aI += __shfl_xor(aI, o, 32);
        aP += __shfl_xor(aP, o, 32);
        aT += __shfl_xor(aT, o, 32);
    }
    if (lane == 0) { pI[tile] = aI; pP[tile] = aP; pT[tile] = aT; }
}

__global__ __launch_bounds__(256) void reduce_k(
    const float* __restrict__ pI, const float* __restrict__ pP,
    const float* __restrict__ pT, float* __restrict__ out, int n)
{
    __shared__ float sI[256], sP[256], sT[256];
    const int tid = threadIdx.x;
    float aI = 0.f, aP = 0.f, aT = 0.f;
    for (int i = tid; i < n; i += 256) { aI += pI[i]; aP += pP[i]; aT += pT[i]; }
    sI[tid] = aI; sP[tid] = aP; sT[tid] = aT;
    __syncthreads();
    for (int s = 128; s; s >>= 1) {
        if (tid < s) {
            sI[tid] += sI[tid + s];
            sP[tid] += sP[tid + s];
            sT[tid] += sT[tid + s];
        }
        __syncthreads();
    }
    if (tid == 0)
        out[0] = 1.0f - (2.0f * sI[0] + 1.0f) / (sP[0] + sT[0] + 1.0f);
}

// ---------------------------------------------------------------- launch
extern "C" void kernel_launch(void* const* d_in, const int* in_sizes, int n_in,
                              void* d_out, int out_size, void* d_ws, size_t ws_size,
                              hipStream_t stream) {
    const float* pred = (const float*)d_in[0];
    const float* tgt  = (const float*)d_in[1];
    float* out = (float*)d_out;

    char* ws = (char*)d_ws;
    unsigned* ctl = (unsigned*)ws;                         // 128 B used
    float* pI = (float*)(ws + 1024);
    float* pP = pI + NTILES;
    float* pT = pP + NTILES;                               // ends < 2 MB
    uint8_t* big   = (uint8_t*)(ws + (2u << 20));
    uint8_t* binP  = big + 0ull * NPIX;
    uint8_t* binT  = big + 1ull * NPIX;
    uint8_t* skPA  = big + 2ull * NPIX;
    uint8_t* skPB  = big + 3ull * NPIX;
    uint8_t* skTA  = big + 4ull * NPIX;
    uint8_t* skTB  = big + 5ull * NPIX;

    init_ctl_k<<<1, 32, 0, stream>>>(ctl);
    binarize_k<<<NPIX / 256, 256, 0, stream>>>(pred, tgt, binP, binT, skPA, skTA);

    for (int it = 0; it < 10; ++it) {
        uint8_t* srcP = (it & 1) ? skPB : skPA;
        uint8_t* dstP = (it & 1) ? skPA : skPB;
        uint8_t* srcT = (it & 1) ? skTB : skTA;
        uint8_t* dstT = (it & 1) ? skTA : skTB;
        erode_k<<<TILE_BLOCKS, 256, 0, stream>>>(srcP, dstP, &ctl[it],      &ctl[20]);
        erode_k<<<TILE_BLOCKS, 256, 0, stream>>>(srcT, dstT, &ctl[10 + it], &ctl[21]);
        flag_k<<<1, 32, 0, stream>>>(ctl, it);
    }
    // 10 iterations (even) -> final skeletons live in the A buffers.

    final_k<<<TILE_BLOCKS, 256, 0, stream>>>(binP, binT, skPA, skTA, pI, pP, pT);
    reduce_k<<<1, 256, 0, stream>>>(pI, pP, pT, out, NTILES);
}
